// DeepQI_24257975288287
// MI455X (gfx1250) — matrix operand order repaired and checked by reading, not compiled, not run on hardware
//
#include <hip/hip_runtime.h>
#include <hip/hip_bf16.h>

#define B_ROWS 16384
#define F_DIM  256
#define NPAIRS 2016
#define H_DIM  2272
#define D1     2048
#define D2     2048
#define D3     1024

typedef __attribute__((ext_vector_type(16))) __bf16 v16bf;
typedef __attribute__((ext_vector_type(8)))  float  v8f;

union FragU { uint4 q[2]; v16bf v; };

__device__ __forceinline__ unsigned short f2bf(float f) {
  unsigned int u = __float_as_uint(f);
  u += 0x7fffu + ((u >> 16) & 1u);        // round-to-nearest-even
  return (unsigned short)(u >> 16);
}
__device__ __forceinline__ float bf2f(unsigned short h) {
  return __uint_as_float(((unsigned int)h) << 16);
}

// ---------------------------------------------------------------- prep kernels

__global__ void DeepQI_make_pairs(int* __restrict__ II, int* __restrict__ JJ) {
  int p = blockIdx.x * blockDim.x + threadIdx.x;
  if (p >= NPAIRS) return;
  int i = 0, acc = 0;
  while (acc + (63 - i) <= p) { acc += 63 - i; ++i; }
  int j = i + 1 + (p - acc);
  II[p] = i; JJ[p] = j;
}

__global__ void DeepQI_cvt_bf16(const float* __restrict__ src,
                                unsigned short* __restrict__ dst, int n) {
  int i = blockIdx.x * blockDim.x + threadIdx.x;
  if (i < n) dst[i] = f2bf(src[i]);
}

__global__ __launch_bounds__(256) void DeepQI_build_h(
    const float* __restrict__ xv, const int* __restrict__ II,
    const int* __restrict__ JJ, unsigned short* __restrict__ hb) {
  __shared__ float xs[F_DIM];
  const int b = blockIdx.x;
  const int t = threadIdx.x;
  xs[t] = xv[(size_t)b * F_DIM + t];
  __syncthreads();
  unsigned short* row = hb + (size_t)b * H_DIM;
  row[t] = f2bf(xs[t]);
  for (int p = t; p < NPAIRS; p += 256)
    row[F_DIM + p] = f2bf(xs[II[p]] * xs[JJ[p]] * (1.0f / 255.0f));
}

// ---------------------------------------------------------------- WMMA GEMM
// Y[M,N] = A[M,K](bf16,row-major) * W[N,K](bf16,row-major)^T + bias[N]
// block tile 128x256, BK=32; 8 waves in 2x4 grid, each wave 64x64 (4x4 wmma
// tiles -> 16 wmma per k-step vs 16 ds_load_b128: 1:1 XDL:LDS issue ratio).

__global__ __launch_bounds__(256) void DeepQI_gemm_bf16(
    const unsigned short* __restrict__ A,
    const unsigned short* __restrict__ W,
    const float* __restrict__ bias,
    float* __restrict__ Y,
    int M, int N, int K)
{
  __shared__ unsigned short Alds[128 * 32];   //  8 KB
  __shared__ unsigned short Blds[256 * 32];   // 16 KB

  const int t    = threadIdx.x;
  const int lane = t & 31;
  const int wave = t >> 5;
  const int wm   = wave >> 2;      // 0..1 : which 64-row slab
  const int wn   = wave & 3;       // 0..3 : which 64-col slab
  const int ln   = lane & 15;
  const int half = lane >> 4;

  const int m0 = blockIdx.x * 128;
  const int n0 = blockIdx.y * 256;

  const int lr = t >> 2;           // 0..63 : tile row for global loads
  const int lc = (t & 3) << 3;     // 0,8,16,24 : element column (16B chunk)

  const v8f zacc = {};
  v8f acc[4][4];
#pragma unroll
  for (int mi = 0; mi < 4; ++mi)
#pragma unroll
    for (int ni = 0; ni < 4; ++ni) acc[mi][ni] = zacc;

  for (int k0 = 0; k0 < K; k0 += 32) {
    const uint4 a0 = *reinterpret_cast<const uint4*>(A + (size_t)(m0 + lr)       * K + k0 + lc);
    const uint4 a1 = *reinterpret_cast<const uint4*>(A + (size_t)(m0 + lr + 64)  * K + k0 + lc);
    const uint4 w0 = *reinterpret_cast<const uint4*>(W + (size_t)(n0 + lr)       * K + k0 + lc);
    const uint4 w1 = *reinterpret_cast<const uint4*>(W + (size_t)(n0 + lr + 64)  * K + k0 + lc);
    const uint4 w2 = *reinterpret_cast<const uint4*>(W + (size_t)(n0 + lr + 128) * K + k0 + lc);
    const uint4 w3 = *reinterpret_cast<const uint4*>(W + (size_t)(n0 + lr + 192) * K + k0 + lc);
    if (k0 + 32 < K) {
      __builtin_prefetch(A + (size_t)(m0 + lr) * K + k0 + 32 + lc, 0, 1);
      __builtin_prefetch(W + (size_t)(n0 + lr) * K + k0 + 32 + lc, 0, 1);
      __builtin_prefetch(W + (size_t)(n0 + lr + 128) * K + k0 + 32 + lc, 0, 1);
    }
    __syncthreads();   // protect previous iteration's LDS reads
    *reinterpret_cast<uint4*>(&Alds[lr * 32 + lc])         = a0;
    *reinterpret_cast<uint4*>(&Alds[(lr + 64) * 32 + lc])  = a1;
    *reinterpret_cast<uint4*>(&Blds[lr * 32 + lc])         = w0;
    *reinterpret_cast<uint4*>(&Blds[(lr + 64) * 32 + lc])  = w1;
    *reinterpret_cast<uint4*>(&Blds[(lr + 128) * 32 + lc]) = w2;
    *reinterpret_cast<uint4*>(&Blds[(lr + 192) * 32 + lc]) = w3;
    __syncthreads();

    // A fragment (16x32 bf16): lane ln holds row m; VGPR v<4 -> k=half*8+2v,
    // VGPR v>=4 -> k=16+half*8+2(v-4): two 16B LDS reads (offset +32B).
    v16bf afrag[4];
#pragma unroll
    for (int mi = 0; mi < 4; ++mi) {
      const int mrow = wm * 64 + mi * 16 + ln;
      const uint4* p = reinterpret_cast<const uint4*>(&Alds[mrow * 32 + half * 8]);
      FragU u; u.q[0] = p[0]; u.q[1] = p[2];
      afrag[mi] = u.v;
    }
    // B fragment (32x16 bf16): lane ln holds col n; same k layout as A:
    // two 16B LDS reads (offset +32B). Loaded in 2 groups to cap live VGPRs.
#pragma unroll
    for (int ng = 0; ng < 2; ++ng) {
      v16bf bfrag[2];
#pragma unroll
      for (int nj = 0; nj < 2; ++nj) {
        const int nrow = wn * 64 + (ng * 2 + nj) * 16 + ln;
        const uint4* p = reinterpret_cast<const uint4*>(&Blds[nrow * 32 + half * 8]);
        FragU u; u.q[0] = p[0]; u.q[1] = p[2];
        bfrag[nj] = u.v;
      }
#pragma unroll
      for (int mi = 0; mi < 4; ++mi)
#pragma unroll
        for (int nj = 0; nj < 2; ++nj)
          acc[mi][ng * 2 + nj] = __builtin_amdgcn_wmma_f32_16x16x32_bf16(
              false, afrag[mi], false, bfrag[nj], (short)0, acc[mi][ng * 2 + nj],
              false, false);
    }
  }

  // C/D layout: lane = (half,ln): n = ln, VGPR r: m = r + 8*half
#pragma unroll
  for (int mi = 0; mi < 4; ++mi) {
    const int gm = m0 + wm * 64 + mi * 16 + half * 8;
#pragma unroll
    for (int ni = 0; ni < 4; ++ni) {
      const int gn = n0 + wn * 64 + ni * 16 + ln;
      const float bb = bias[gn];
#pragma unroll
      for (int r = 0; r < 8; ++r)
        Y[(size_t)(gm + r) * N + gn] = acc[mi][ni][r] + bb;
    }
  }
}

// ---------------------------------------------------------------- batch-norm

__global__ void DeepQI_zero_f32(float* __restrict__ p, int n) {
  int i = blockIdx.x * blockDim.x + threadIdx.x;
  if (i < n) p[i] = 0.f;
}

__global__ __launch_bounds__(256) void DeepQI_colstats(
    const float* __restrict__ Y, float* __restrict__ sum,
    float* __restrict__ sumsq, int M, int N)
{
  __shared__ float ss[256], sq[256];
  const int t  = threadIdx.x;
  const int c  = blockIdx.x * 64 + (t & 63);
  const int ty = t >> 6;                    // 0..3
  const int rows = M / gridDim.y;
  const int r0 = blockIdx.y * rows;
  float s = 0.f, q = 0.f;
  for (int r = r0 + ty; r < r0 + rows; r += 4) {
    float v = Y[(size_t)r * N + c];
    s += v; q += v * v;
  }
  ss[t] = s; sq[t] = q;
  __syncthreads();
  if (ty == 0) {
    s = ss[t] + ss[t + 64] + ss[t + 128] + ss[t + 192];
    q = sq[t] + sq[t + 64] + sq[t + 128] + sq[t + 192];
    atomicAdd(&sum[c], s);
    atomicAdd(&sumsq[c], q);
  }
}

__global__ void DeepQI_finalize_bn(const float* __restrict__ sum,
                                   const float* __restrict__ sumsq,
                                   const float* __restrict__ g,
                                   float* __restrict__ mean,
                                   float* __restrict__ scale,
                                   int N, float invM) {
  int c = blockIdx.x * blockDim.x + threadIdx.x;
  if (c >= N) return;
  float m = sum[c] * invM;
  float v = sumsq[c] * invM - m * m;
  mean[c]  = m;
  scale[c] = g[c] * rsqrtf(v + 1e-5f);
}

// grid: (N/256, 256); each thread owns one column slot, loops rows.
__global__ __launch_bounds__(256) void DeepQI_bn_relu_bf16(
    const float* __restrict__ Y, const float* __restrict__ mean,
    const float* __restrict__ scale, const float* __restrict__ beta,
    unsigned short* __restrict__ act, int M, int N) {
  const int c = blockIdx.x * 256 + threadIdx.x;
  const float mn = mean[c], sc = scale[c], bt = beta[c];
  for (int r = blockIdx.y; r < M; r += gridDim.y) {
    const size_t i = (size_t)r * N + c;
    float x = (Y[i] - mn) * sc + bt;
    act[i] = f2bf(x > 0.f ? x : 0.f);
  }
}

// ---------------------------------------------------------------- output head

__global__ __launch_bounds__(256) void DeepQI_out(
    const unsigned short* __restrict__ act,
    const unsigned short* __restrict__ wout,
    const float* __restrict__ bout,
    float* __restrict__ out, int M) {
  const int wave = threadIdx.x >> 5;
  const int lane = threadIdx.x & 31;
  const int row  = blockIdx.x * 8 + wave;
  if (row >= M) return;
  const unsigned short* a = act + (size_t)row * D3;
  float s = 0.f;
  for (int k = lane; k < D3; k += 32) s += bf2f(a[k]) * bf2f(wout[k]);
#pragma unroll
  for (int off = 16; off > 0; off >>= 1) s += __shfl_down(s, off, 32);
  if (lane == 0) out[row] = s + bout[0];
}

// ---------------------------------------------------------------- launch

extern "C" void kernel_launch(void* const* d_in, const int* in_sizes, int n_in,
                              void* d_out, int out_size, void* d_ws, size_t ws_size,
                              hipStream_t stream)
{
  (void)in_sizes; (void)n_in; (void)out_size; (void)ws_size;
  const float* xv   = (const float*)d_in[0];
  const float* W0   = (const float*)d_in[1];
  const float* b0   = (const float*)d_in[2];
  const float* g0   = (const float*)d_in[3];
  const float* be0  = (const float*)d_in[4];
  const float* W1   = (const float*)d_in[5];
  const float* b1   = (const float*)d_in[6];
  const float* g1   = (const float*)d_in[7];
  const float* be1  = (const float*)d_in[8];
  const float* W2   = (const float*)d_in[9];
  const float* b2   = (const float*)d_in[10];
  const float* g2   = (const float*)d_in[11];
  const float* be2  = (const float*)d_in[12];
  const float* Wout = (const float*)d_in[13];
  const float* bout = (const float*)d_in[14];
  float* out = (float*)d_out;

  unsigned char* base = (unsigned char*)d_ws;
  size_t off = 0;
  auto take = [&](size_t bytes) -> void* {
    size_t o = (off + 255) & ~(size_t)255;
    off = o + bytes;
    return (void*)(base + o);
  };

  unsigned short* hb  = (unsigned short*)take((size_t)B_ROWS * H_DIM * 2);
  unsigned short* w0b = (unsigned short*)take((size_t)D1 * H_DIM * 2);
  unsigned short* w1b = (unsigned short*)take((size_t)D2 * D1 * 2);
  unsigned short* w2b = (unsigned short*)take((size_t)D3 * D2 * 2);
  unsigned short* wob = (unsigned short*)take((size_t)D3 * 2);
  float*          Y   = (float*)take((size_t)B_ROWS * D1 * 4);          // reused per layer
  unsigned short* act = (unsigned short*)take((size_t)B_ROWS * D1 * 2); // reused per layer
  int*   II    = (int*)take((size_t)NPAIRS * 4);
  int*   JJ    = (int*)take((size_t)NPAIRS * 4);
  float* stats = (float*)take((size_t)4 * D1 * 4);
  float* sum   = stats;
  float* sumsq = stats + D1;
  float* mean  = stats + 2 * D1;
  float* scale = stats + 3 * D1;

  // --- prep
  DeepQI_make_pairs<<<(NPAIRS + 255) / 256, 256, 0, stream>>>(II, JJ);
  DeepQI_cvt_bf16<<<(D1 * H_DIM + 255) / 256, 256, 0, stream>>>(W0, w0b, D1 * H_DIM);
  DeepQI_cvt_bf16<<<(D2 * D1 + 255) / 256, 256, 0, stream>>>(W1, w1b, D2 * D1);
  DeepQI_cvt_bf16<<<(D3 * D2 + 255) / 256, 256, 0, stream>>>(W2, w2b, D3 * D2);
  DeepQI_cvt_bf16<<<(D3 + 255) / 256, 256, 0, stream>>>(Wout, wob, D3);
  DeepQI_build_h<<<B_ROWS, 256, 0, stream>>>(xv, II, JJ, hb);

  auto run_layer = [&](const unsigned short* Ain, const unsigned short* Wb,
                       const float* bias, const float* g, const float* be,
                       int N, int K) {
    DeepQI_zero_f32<<<(2 * D1 + 255) / 256, 256, 0, stream>>>(stats, 2 * D1);
    dim3 ggrid(B_ROWS / 128, N / 256);
    DeepQI_gemm_bf16<<<ggrid, 256, 0, stream>>>(Ain, Wb, bias, Y, B_ROWS, N, K);
    dim3 sgrid(N / 64, 32);
    DeepQI_colstats<<<sgrid, 256, 0, stream>>>(Y, sum, sumsq, B_ROWS, N);
    DeepQI_finalize_bn<<<(N + 255) / 256, 256, 0, stream>>>(
        sum, sumsq, g, mean, scale, N, 1.0f / (float)B_ROWS);
    dim3 ngrid(N / 256, 256);
    DeepQI_bn_relu_bf16<<<ngrid, 256, 0, stream>>>(Y, mean, scale, be, act, B_ROWS, N);
  };

  run_layer(hb,  w0b, b0, g0, be0, D1, H_DIM);   // 16384 x 2272 -> 2048
  run_layer(act, w1b, b1, g1, be1, D2, D1);      // 16384 x 2048 -> 2048
  run_layer(act, w2b, b2, g2, be2, D3, D2);      // 16384 x 2048 -> 1024

  DeepQI_out<<<B_ROWS / 8, 256, 0, stream>>>(act, wob, bout, out, B_ROWS);
}
